// Head_7146825580982
// MI455X (gfx1250) — matrix operand order repaired and checked by reading, not compiled, not run on hardware
//
#include <hip/hip_runtime.h>

#define BATCH 8
#define T_SEQ 2048
#define CDIM  768
#define HDIM  64

typedef __attribute__((ext_vector_type(16))) __bf16 v16bf;
typedef __attribute__((ext_vector_type(8)))  __bf16 bf16x8;
typedef __attribute__((ext_vector_type(8)))  float  v8f;
typedef __attribute__((ext_vector_type(4)))  float  f32x4;
typedef int v4i_g __attribute__((__vector_size__(4 * sizeof(int))));   // async builtin param
typedef unsigned int u32x4 __attribute__((__vector_size__(16)));       // TDM D# group 0
typedef int i32x8 __attribute__((__vector_size__(32)));                // TDM D# group 1
typedef int i32x4 __attribute__((__vector_size__(16)));                // TDM D# groups 2/3

#define AS1 __attribute__((address_space(1)))
#define AS3 __attribute__((address_space(3)))

// ---- workspace byte offsets ----
#define WPACK_OFF 0u
#define QBF_OFF   294912u
#define KBF_OFF   (294912u + 2097152u)
#define VT_OFF    (294912u + 2u * 2097152u)

// ---- gfx1250 async-to-LDS path (probe-verified present on this toolchain) ----
#if defined(__gfx1250__) && __has_builtin(__builtin_amdgcn_global_load_async_to_lds_b128)
#define HAVE_ASYNC_LDS 1
#define ASYNC_CP_B128(g, l) \
  __builtin_amdgcn_global_load_async_to_lds_b128((AS1 v4i_g*)(g), (AS3 v4i_g*)(l), 0, 0)
#if __has_builtin(__builtin_amdgcn_s_wait_asynccnt)
#define WAIT_ASYNC() __builtin_amdgcn_s_wait_asynccnt(0)
#else
#define WAIT_ASYNC() asm volatile("s_wait_asynccnt 0" ::: "memory")
#endif
#else
#define HAVE_ASYNC_LDS 0
#define WAIT_ASYNC() ((void)0)
#endif

// ---- gfx1250 Tensor Data Mover path (guarded; arity differs per toolchain) ----
#if defined(__gfx1250__) && __has_builtin(__builtin_amdgcn_tensor_load_to_lds)
#define HAVE_TDM 1
#if __has_builtin(__builtin_amdgcn_s_wait_tensorcnt)
#define WAIT_TENSOR() __builtin_amdgcn_s_wait_tensorcnt(0)
#else
#define WAIT_TENSOR() asm volatile("s_wait_tensorcnt 0" ::: "memory")
#endif

// 2D TDM tile load: tensor[tensor_d1][tensor_d0] (2-byte elems, row stride d0_stride
// elems) -> compacted LDS tile [tile_d1][tile_d0]. D# layout per CDNA5 ISA ch.8.
__device__ __forceinline__ void tdm_load_2d(const void* gptr, void* lptr,
                                            unsigned tensor_d0, unsigned tensor_d1,
                                            unsigned d0_stride,
                                            unsigned tile_d0, unsigned tile_d1)
{
  unsigned long long ga = (unsigned long long)(__SIZE_TYPE__)gptr;
  unsigned lds = (unsigned)(__SIZE_TYPE__)(AS3 char*)lptr;   // LDS byte address
  u32x4 g0 = {};
  g0[0] = 1u;                                        // count=1 (valid), user mode
  g0[1] = lds;                                       // lds_addr
  g0[2] = (unsigned)(ga & 0xffffffffu);              // global_addr[31:0]
  g0[3] = (unsigned)((ga >> 32) & 0x1ffffffu) | (2u << 30);  // addr[56:32] | type=2
  i32x8 g1 = {};
  g1[0] = (int)(1u << 16);                           // data_size=1 -> 2 bytes
  g1[1] = (int)(tensor_d0 << 16);                    // tensor_dim0[15:0]
  g1[2] = (int)((tensor_d0 >> 16) | (tensor_d1 << 16));
  g1[3] = (int)((tensor_d1 >> 16) | (tile_d0 << 16));
  g1[4] = (int)(tile_d1 & 0xffffu);                  // tile_dim1 (tile_dim2=0)
  g1[5] = (int)d0_stride;                            // tensor_dim0_stride[31:0]
  i32x4 z4 = {};
#if __clang_major__ >= 23
  i32x8 z8 = {};
  __builtin_amdgcn_tensor_load_to_lds(g0, g1, z4, z4, z8, 0);
#else
  __builtin_amdgcn_tensor_load_to_lds(g0, g1, z4, z4, 0);
#endif
}
#else
#define HAVE_TDM 0
#define WAIT_TENSOR() ((void)0)
#endif

// -------------------------------------------------------------------------
// Kernel 1: pack W into WMMA B-operand layout, bf16.
// -------------------------------------------------------------------------
__global__ __launch_bounds__(256) void pack_w_kernel(const float* __restrict__ Wq,
                                                     const float* __restrict__ Wk,
                                                     const float* __restrict__ Wv,
                                                     __bf16* __restrict__ wp)
{
  int gid = blockIdx.x * 256 + threadIdx.x;
  if (gid >= 3 * 49152) return;
  int m  = gid / 49152;
  int r  = gid - m * 49152;
  int e  = r & 15;
  int L  = (r >> 4) & 31;
  int j  = (r >> 9) & 3;
  int kc = r >> 11;
  int row = kc * 32 + (L >> 4) * 8 + (e & 7) + (e >> 3) * 16;
  int col = j * 16 + (L & 15);
  const float* W = (m == 0) ? Wq : ((m == 1) ? Wk : Wv);
  wp[gid] = (__bf16)W[row * HDIM + col];
}

// -------------------------------------------------------------------------
// Kernel 2: fused QKV projection (one wave = 16 rows x 64 cols of Q,K,V).
// -------------------------------------------------------------------------
__global__ __launch_bounds__(256) void proj_qkv_kernel(const float* __restrict__ x,
                                                       const __bf16* __restrict__ wp,
                                                       __bf16* __restrict__ qbf,
                                                       __bf16* __restrict__ kbf,
                                                       __bf16* __restrict__ vt)
{
  const int w    = threadIdx.x >> 5;
  const int lane = threadIdx.x & 31;
  const int hf   = lane >> 4;
  const int lm   = lane & 15;
  const int rowbase = blockIdx.x * 128 + w * 16;
  const float* xrow = x + (size_t)(rowbase + lm) * CDIM;

  v8f zero = {};
  v8f acc[3][4];
#pragma unroll
  for (int m = 0; m < 3; ++m)
#pragma unroll
    for (int j = 0; j < 4; ++j) acc[m][j] = zero;

  for (int kc = 0; kc < 24; ++kc) {
    const float* p0 = xrow + kc * 32 + hf * 8;
    f32x4 a0 = *(const f32x4*)(p0);
    f32x4 a1 = *(const f32x4*)(p0 + 4);
    f32x4 a2 = *(const f32x4*)(p0 + 16);
    f32x4 a3 = *(const f32x4*)(p0 + 20);
    v16bf a;
#pragma unroll
    for (int e = 0; e < 4; ++e) {
      a[e]      = (__bf16)a0[e];
      a[e + 4]  = (__bf16)a1[e];
      a[e + 8]  = (__bf16)a2[e];
      a[e + 12] = (__bf16)a3[e];
    }
#pragma unroll
    for (int m = 0; m < 3; ++m) {
      v16bf bt[4];
#pragma unroll
      for (int j = 0; j < 4; ++j) {
        const __bf16* bp = wp + m * 49152 + ((kc * 4 + j) * 32 + lane) * 16;
        bf16x8 b0 = *(const bf16x8*)(bp);
        bf16x8 b1 = *(const bf16x8*)(bp + 8);
#pragma unroll
        for (int e = 0; e < 8; ++e) { bt[j][e] = b0[e]; bt[j][e + 8] = b1[e]; }
      }
#pragma unroll
      for (int j = 0; j < 4; ++j)
        acc[m][j] = __builtin_amdgcn_wmma_f32_16x16x32_bf16(
            false, a, false, bt[j], (short)0, acc[m][j], false, false);
    }
  }

  const int b = rowbase >> 11;
#pragma unroll
  for (int j = 0; j < 4; ++j) {
    const int col = j * 16 + lm;
#pragma unroll
    for (int i = 0; i < 8; ++i) {
      const int R = rowbase + hf * 8 + i;
      const size_t rq = (size_t)R * HDIM + col;
      qbf[rq] = (__bf16)(acc[0][j][i] * 0.125f);  // fold 1/sqrt(64) into Q
      kbf[rq] = (__bf16)acc[1][j][i];
      vt[((size_t)b * HDIM + col) * T_SEQ + (R & (T_SEQ - 1))] = (__bf16)acc[2][j][i];
    }
  }
}

// -------------------------------------------------------------------------
// Kernel 3: flash attention. K staged via async-to-LDS (contiguous 4KB),
// V^T staged via TDM tensor_load_to_lds (strided 2D tile), double-buffered.
// -------------------------------------------------------------------------
__global__ __launch_bounds__(256) void attn_fwd_kernel(const __bf16* __restrict__ qbf,
                                                       const __bf16* __restrict__ kbf,
                                                       const __bf16* __restrict__ vt,
                                                       float* __restrict__ out)
{
  __shared__ __align__(16) __bf16 kbuf[2][32 * 64];  // K block, row-major [s][h]
  __shared__ __align__(16) __bf16 vbuf[2][64 * 32];  // V^T block, [h][s]
  __shared__ __align__(16) __bf16 pshm[8][16 * 32];  // per-wave P relay (C->A layout)

  const int tid  = threadIdx.x;
  const int w    = tid >> 5;
  const int lane = tid & 31;
  const int hf   = lane >> 4;
  const int lm   = lane & 15;

  const int gw    = blockIdx.x * 8 + w;
  const int b     = gw >> 7;
  const int qbase = (gw & 127) * 16;
  const int smax  = ((blockIdx.x & 15) * 8 + 7) * 16;  // wave 7's qbase (block max)

  const __bf16* kbase = kbf + (size_t)b * T_SEQ * HDIM;
  const __bf16* vbase = vt  + (size_t)b * HDIM * T_SEQ;
  __bf16* myp = &pshm[w][0];

  // Stage one 32-key block: K = 16B/thread async copies; V^T = one TDM descriptor.
  auto stage = [&](int sb2, int buf) {
    const __bf16* kg = kbase + (size_t)sb2 * HDIM + tid * 8;   // contiguous 4KB
    __bf16* kl = &kbuf[buf][tid * 8];
#if HAVE_ASYNC_LDS
    ASYNC_CP_B128(kg, kl);
#else
    *(bf16x8*)kl = *(const bf16x8*)kg;
#endif
#if HAVE_TDM
    if (w == 0)  // wave-uniform; TDM ignores EXEC, one DMA covers the whole tile
      tdm_load_2d(vbase + sb2, &vbuf[buf][0],
                  /*tensor_d0=*/T_SEQ, /*tensor_d1=*/HDIM, /*d0_stride=*/T_SEQ,
                  /*tile_d0=*/32, /*tile_d1=*/HDIM);
#elif HAVE_ASYNC_LDS
    const __bf16* vg = vbase + (size_t)(tid >> 2) * T_SEQ + sb2 + (tid & 3) * 8;
    ASYNC_CP_B128(vg, &vbuf[buf][tid * 8]);
#else
    const __bf16* vg = vbase + (size_t)(tid >> 2) * T_SEQ + sb2 + (tid & 3) * 8;
    *(bf16x8*)&vbuf[buf][tid * 8] = *(const bf16x8*)vg;
#endif
  };

  // Q tile in A-layout (both 32-wide H chunks), resident for the whole kernel.
  v16bf qa0, qa1;
  {
    const __bf16* qp = qbf + (size_t)(b * T_SEQ + qbase + lm) * HDIM;
    bf16x8 r0 = *(const bf16x8*)(qp + hf * 8);
    bf16x8 r1 = *(const bf16x8*)(qp + 16 + hf * 8);
    bf16x8 r2 = *(const bf16x8*)(qp + 32 + hf * 8);
    bf16x8 r3 = *(const bf16x8*)(qp + 48 + hf * 8);
#pragma unroll
    for (int e = 0; e < 8; ++e) {
      qa0[e] = r0[e]; qa0[e + 8] = r1[e];
      qa1[e] = r2[e]; qa1[e + 8] = r3[e];
    }
  }

  v8f zero = {};
  v8f o[4] = {zero, zero, zero, zero};
  float mi[8], li[8];
#pragma unroll
  for (int i = 0; i < 8; ++i) { mi[i] = -1e30f; li[i] = 0.0f; }

  stage(0, 0);
  WAIT_ASYNC();
  WAIT_TENSOR();
  __syncthreads();

  for (int sb = 0; sb <= smax; sb += 32) {
    const int cur = (sb >> 5) & 1;
    if (sb + 32 <= smax) stage(sb + 32, cur ^ 1);   // prefetch next block into other buffer

    if (sb <= qbase) {   // wave-uniform: causal range of this wave's q tile
      const __bf16* kl = &kbuf[cur][0];
      const __bf16* vl = &vbuf[cur][0];

      // --- S = Q K^T : load all 4 K B-tiles, then 4 WMMAs ---
      v16bf kt[4];
#pragma unroll
      for (int t = 0; t < 2; ++t)
#pragma unroll
        for (int c = 0; c < 2; ++c) {
          const __bf16* kp = kl + (t * 16 + lm) * 64 + c * 32 + hf * 8;
          bf16x8 r0 = *(const bf16x8*)(kp);
          bf16x8 r1 = *(const bf16x8*)(kp + 16);
#pragma unroll
          for (int e = 0; e < 8; ++e) { kt[t * 2 + c][e] = r0[e]; kt[t * 2 + c][e + 8] = r1[e]; }
        }
      v8f s0 = zero, s1 = zero;
      s0 = __builtin_amdgcn_wmma_f32_16x16x32_bf16(false, qa0, false, kt[0], (short)0, s0, false, false);
      s0 = __builtin_amdgcn_wmma_f32_16x16x32_bf16(false, qa1, false, kt[1], (short)0, s0, false, false);
      s1 = __builtin_amdgcn_wmma_f32_16x16x32_bf16(false, qa0, false, kt[2], (short)0, s1, false, false);
      s1 = __builtin_amdgcn_wmma_f32_16x16x32_bf16(false, qa1, false, kt[3], (short)0, s1, false, false);

      // --- V B-tiles issued early (independent of softmax) ---
      v16bf vtile[4];
#pragma unroll
      for (int j = 0; j < 4; ++j) {
        const __bf16* vp = vl + (j * 16 + lm) * 32 + hf * 8;
        bf16x8 va = *(const bf16x8*)(vp);
        bf16x8 vb = *(const bf16x8*)(vp + 16);
#pragma unroll
        for (int e = 0; e < 8; ++e) { vtile[j][e] = va[e]; vtile[j][e + 8] = vb[e]; }
      }

      if (sb + 32 > qbase) {  // uniform: only the diagonal block is masked
#pragma unroll
        for (int i = 0; i < 8; ++i) {
          const int mg = qbase + hf * 8 + i;
          s0[i] = (sb + lm      > mg) ? -1e30f : s0[i];
          s1[i] = (sb + 16 + lm > mg) ? -1e30f : s1[i];
        }
      }

      // --- online softmax; C layout: VGPR i = row (i + hf*8) across 16 lanes ---
#pragma unroll
      for (int i = 0; i < 8; ++i) {
        float rmax = fmaxf(s0[i], s1[i]);
#pragma unroll
        for (int d = 1; d < 16; d <<= 1) rmax = fmaxf(rmax, __shfl_xor(rmax, d, 32));
        const float mnew = fmaxf(mi[i], rmax);
        const float corr = exp2f((mi[i] - mnew) * 1.4426950408889634f);
        const float e0   = exp2f((s0[i] - mnew) * 1.4426950408889634f);
        const float e1   = exp2f((s1[i] - mnew) * 1.4426950408889634f);
        float rs = e0 + e1;
#pragma unroll
        for (int d = 1; d < 16; d <<= 1) rs += __shfl_xor(rs, d, 32);
        li[i] = li[i] * corr + rs;
        mi[i] = mnew;
        o[0][i] *= corr; o[1][i] *= corr; o[2][i] *= corr; o[3][i] *= corr;
        myp[(hf * 8 + i) * 32 + lm]      = (__bf16)e0;
        myp[(hf * 8 + i) * 32 + 16 + lm] = (__bf16)e1;
      }

      // --- re-read P in A layout (same-wave DS ops are in-order) ---
      v16bf pa;
      {
        bf16x8 r0 = *(const bf16x8*)(myp + lm * 32 + hf * 8);
        bf16x8 r1 = *(const bf16x8*)(myp + lm * 32 + 16 + hf * 8);
#pragma unroll
        for (int e = 0; e < 8; ++e) { pa[e] = r0[e]; pa[e + 8] = r1[e]; }
      }

      // --- O += P V ---
#pragma unroll
      for (int j = 0; j < 4; ++j)
        o[j] = __builtin_amdgcn_wmma_f32_16x16x32_bf16(
            false, pa, false, vtile[j], (short)0, o[j], false, false);
    }

    WAIT_ASYNC();      // my async copies for the next buffer landed in LDS
    WAIT_TENSOR();     // wave 0's TDM tile landed too (no-op counter for others)
    __syncthreads();   // block-wide visibility + WAR close on the consumed buffer
  }

#pragma unroll
  for (int i = 0; i < 8; ++i) {
    const float inv = 1.0f / li[i];
    const int m = qbase + hf * 8 + i;
    float* op = out + (size_t)(b * T_SEQ + m) * HDIM + lm;
#pragma unroll
    for (int j = 0; j < 4; ++j) op[j * 16] = o[j][i] * inv;
  }
}

// -------------------------------------------------------------------------
extern "C" void kernel_launch(void* const* d_in, const int* in_sizes, int n_in,
                              void* d_out, int out_size, void* d_ws, size_t ws_size,
                              hipStream_t stream)
{
  (void)in_sizes; (void)n_in; (void)out_size; (void)ws_size;
  const float* x  = (const float*)d_in[0];
  const float* Wk = (const float*)d_in[1];
  const float* Wq = (const float*)d_in[2];
  const float* Wv = (const float*)d_in[3];
  float* out = (float*)d_out;

  char* ws = (char*)d_ws;
  __bf16* wp  = (__bf16*)(ws + WPACK_OFF);
  __bf16* qbf = (__bf16*)(ws + QBF_OFF);
  __bf16* kbf = (__bf16*)(ws + KBF_OFF);
  __bf16* vt  = (__bf16*)(ws + VT_OFF);

  pack_w_kernel<<<576, 256, 0, stream>>>(Wq, Wk, Wv, wp);
  proj_qkv_kernel<<<128, 256, 0, stream>>>(x, wp, qbf, kbf, vt);
  attn_fwd_kernel<<<128, 256, 0, stream>>>(qbf, kbf, vt, out);
}
